// MinEuclideanDistBlockACS_69166153335418
// MI455X (gfx1250) — matrix operand order repaired and checked by reading, not compiled, not run on hardware
//
#include <hip/hip_runtime.h>
#include <math.h>

// Problem constants (from reference)
#define N_    64
#define C_    3
#define L_    4096
#define K_    128
#define S_    64
#define W_    4033      // L - S + 1 valid windows
#define WPAD_ 4048      // 253 * 16 (window dim padded to tile multiple)
#define NTILES_ 253
#define LXPAD_ 4224     // padded x row: covers max read 4110, zero tail

typedef __bf16 bf16x16 __attribute__((ext_vector_type(16)));
typedef float  f32x8   __attribute__((ext_vector_type(8)));

static __device__ __forceinline__ unsigned short f2bf(float f) {
  // round-to-nearest-even fp32 -> bf16 (inputs are finite gaussians; no NaN path needed)
  unsigned u = __float_as_uint(f);
  u += 0x7FFFu + ((u >> 16) & 1u);
  return (unsigned short)(u >> 16);
}
static __device__ __forceinline__ float bf2f(unsigned short h) {
  return __uint_as_float(((unsigned)h) << 16);
}
static __device__ __forceinline__ __bf16 us2bf(unsigned short h) {
  union { unsigned short u; __bf16 b; } cv; cv.u = h; return cv.b;
}

// ---------------- preprocessing kernels ----------------

// x -> bf16 hi/lo split, padded to LXPAD_ with zeros
__global__ void cvt_x_kernel(const float* __restrict__ x,
                             unsigned short* __restrict__ xh,
                             unsigned short* __restrict__ xl) {
  int idx = blockIdx.x * blockDim.x + threadIdx.x;
  if (idx >= N_ * C_ * LXPAD_) return;
  int i  = idx % LXPAD_;
  int nc = idx / LXPAD_;
  float v = (i < L_) ? x[nc * L_ + i] : 0.0f;
  unsigned short h = f2bf(v);
  xh[idx] = h;
  xl[idx] = f2bf(v - bf2f(h));
}

// shapelets -> bf16 hi/lo split + exact fp32 sum-of-squares per shapelet
__global__ void cvt_s_kernel(const float* __restrict__ s,
                             unsigned short* __restrict__ sh,
                             unsigned short* __restrict__ sl,
                             float* __restrict__ s2) {
  int t = blockIdx.x * blockDim.x + threadIdx.x;
  if (t >= C_ * K_) return;
  const float* row = s + t * S_;
  float acc = 0.0f;
  for (int j = 0; j < S_; ++j) {
    float v = row[j];
    acc += v * v;
    unsigned short h = f2bf(v);
    sh[t * S_ + j] = h;
    sl[t * S_ + j] = f2bf(v - bf2f(h));
  }
  s2[t] = acc;
}

// sliding-window ||x_w||^2 in exact fp32; +INF padding masks invalid windows
__global__ void x2win_kernel(const float* __restrict__ x,
                             float* __restrict__ x2w) {
  int idx = blockIdx.x * blockDim.x + threadIdx.x;
  if (idx >= N_ * C_ * WPAD_) return;
  int w  = idx % WPAD_;
  int nc = idx / WPAD_;
  float r;
  if (w < W_) {
    const float* p = x + nc * L_ + w;
    float acc = 0.0f;
#pragma unroll 8
    for (int j = 0; j < S_; ++j) { float v = p[j]; acc += v * v; }
    r = acc;
  } else {
    r = __builtin_inff();
  }
  x2w[idx] = r;
}

// ---------------- main WMMA kernel ----------------
// grid = (N, 4 k-tile-pairs, C), block = 128 (4 waves); wave w: tiles t = w, w+4, ...
// Each block covers TWO 16-shapelet tiles -> two independent accumulator chains:
// every A fragment feeds 12 WMMAs (2x compute per LDS byte) and alternating
// acc0/acc1 issue order keeps an independent WMMA between every dependent pair
// (fills the WMMA->WMMA / WMMA->VALU hazard slots).
// Dots use the bf16 hi/lo split: hi*hi + hi*lo + lo*hi ~ fp32 accuracy.
// Running min is kept in SQUARED-distance space (sqrt is monotone); the single
// clamp+sqrt happens in reduce_kernel.
__global__ void __launch_bounds__(128)
shapelet_main_kernel(const unsigned short* __restrict__ xh,
                     const unsigned short* __restrict__ xl,
                     const unsigned short* __restrict__ sh,
                     const unsigned short* __restrict__ sl,
                     const float* __restrict__ s2,
                     const float* __restrict__ x2w,
                     float* __restrict__ partial) {
  const int n   = blockIdx.x;
  const int ktp = blockIdx.y;  // pair of 16-shapelet tiles: kt = 2*ktp, 2*ktp+1
  const int c   = blockIdx.z;
  const int tid  = threadIdx.x;
  const int wave = tid >> 5;
  const int lane = tid & 31;

  // Stage the whole (n,c) x row (hi+lo) in LDS once; all 4 waves reuse it.
  __shared__ unsigned short sxh[LXPAD_];
  __shared__ unsigned short sxl[LXPAD_];
  {
    const unsigned short* gh = xh + (size_t)(n * C_ + c) * LXPAD_;
    const unsigned short* gl = xl + (size_t)(n * C_ + c) * LXPAD_;
    for (int i = tid; i < LXPAD_; i += 128) { sxh[i] = gh[i]; sxl[i] = gl[i]; }
  }
  __syncthreads();

  // Lane roles as used here (16x16x32 bf16, wave32; A/B k order is this code's contiguous choice):
  //  A: lane&15 = row M; lanes<16 hold K 0..15, lanes>=16 hold K 16..31
  //  B: lane&15 = col N; lanes<16 hold K 0..15, lanes>=16 hold K 16..31
  //  C/D: vgpr i -> M = i (+8 for upper half-wave), N = lane&15
  const int col   = lane & 15;
  const int kb    = (lane & 16) ? 16 : 0;  // B k-offset
  const int koff  = (lane & 16) ? 16 : 0;  // A k-offset
  const int mbase = (lane & 16) ? 8  : 0;  // C/D row base

  // B fragments for both sub-tiles: [h][kstep], hi and lo (32B-aligned loads)
  bf16x16 bh[2][2], bl[2][2];
#pragma unroll
  for (int h = 0; h < 2; ++h) {
    const unsigned short* brh = sh + (size_t)(c * K_ + ktp * 32 + h * 16 + col) * S_;
    const unsigned short* brl = sl + (size_t)(c * K_ + ktp * 32 + h * 16 + col) * S_;
    bh[h][0] = *(const bf16x16*)(brh + 0  + kb);
    bh[h][1] = *(const bf16x16*)(brh + 32 + kb);
    bl[h][0] = *(const bf16x16*)(brl + 0  + kb);
    bl[h][1] = *(const bf16x16*)(brl + 32 + kb);
  }

  const float s2v0 = s2[c * K_ + ktp * 32 + col];
  const float s2v1 = s2[c * K_ + ktp * 32 + 16 + col];
  const float* x2row = x2w + (size_t)(n * C_ + c) * WPAD_;

  float lmin0 = __builtin_inff();  // running min of squared distance, sub-tile 0
  float lmin1 = __builtin_inff();  // sub-tile 1

  for (int t = wave; t < NTILES_; t += 4) {
    const int w0 = t * 16;
    const int abase = w0 + col;  // row M = lane&15 (Toeplitz: row w reads x[w..w+63])

    f32x8 acc0, acc1;
#pragma unroll
    for (int i = 0; i < 8; ++i) { acc0[i] = 0.0f; acc1[i] = 0.0f; }

#pragma unroll
    for (int ks = 0; ks < 2; ++ks) {
      const int kk = abase + ks * 32 + koff;
      bf16x16 ah, al;
#pragma unroll
      for (int j = 0; j < 8; ++j) {
        ah[j]     = us2bf(sxh[kk + j]);
        ah[j + 8] = us2bf(sxh[kk + 8 + j]);
        al[j]     = us2bf(sxl[kk + j]);
        al[j + 8] = us2bf(sxl[kk + 8 + j]);
      }
      // interleave the two independent chains
      acc0 = __builtin_amdgcn_wmma_f32_16x16x32_bf16(false, ah, false, bh[0][ks], (short)0, acc0, false, false);
      acc1 = __builtin_amdgcn_wmma_f32_16x16x32_bf16(false, ah, false, bh[1][ks], (short)0, acc1, false, false);
      acc0 = __builtin_amdgcn_wmma_f32_16x16x32_bf16(false, ah, false, bl[0][ks], (short)0, acc0, false, false);
      acc1 = __builtin_amdgcn_wmma_f32_16x16x32_bf16(false, ah, false, bl[1][ks], (short)0, acc1, false, false);
      acc0 = __builtin_amdgcn_wmma_f32_16x16x32_bf16(false, al, false, bh[0][ks], (short)0, acc0, false, false);
      acc1 = __builtin_amdgcn_wmma_f32_16x16x32_bf16(false, al, false, bh[1][ks], (short)0, acc1, false, false);
    }

    // epilogue: d2 = ||x_w||^2 + ||s||^2 - 2*dot; track min of d2 only.
    // (padded windows carry +INF in x2row and drop out of the min automatically)
#pragma unroll
    for (int i = 0; i < 8; ++i) {
      float xw = x2row[w0 + mbase + i];
      lmin0 = fminf(lmin0, xw + s2v0 - 2.0f * acc0[i]);
      lmin1 = fminf(lmin1, xw + s2v1 - 2.0f * acc1[i]);
    }
  }

  // fold the two M-halves (same shapelet column lives in lane and lane^16)
  lmin0 = fminf(lmin0, __shfl_xor(lmin0, 16, 32));
  lmin1 = fminf(lmin1, __shfl_xor(lmin1, 16, 32));
  if (lane < 16) {
    const size_t base0 = (size_t)(((n * 8 + ktp * 2 + 0) * C_ + c) * 4 + wave) * 16 + col;
    const size_t base1 = (size_t)(((n * 8 + ktp * 2 + 1) * C_ + c) * 4 + wave) * 16 + col;
    partial[base0] = lmin0;
    partial[base1] = lmin1;
  }
}

// min over channels and window-chunks, then the single clamp+sqrt -> out (N,1,K)
__global__ void reduce_kernel(const float* __restrict__ partial,
                              float* __restrict__ out) {
  int n = blockIdx.x;
  int k = threadIdx.x;          // 128
  int kt = k >> 4, col = k & 15;
  float m = __builtin_inff();
#pragma unroll
  for (int c = 0; c < C_; ++c)
#pragma unroll
    for (int ch = 0; ch < 4; ++ch)
      m = fminf(m, partial[(size_t)(((n * 8 + kt) * C_ + c) * 4 + ch) * 16 + col]);
  m = m > 0.0f ? m : 0.0f;
  out[n * K_ + k] = __builtin_sqrtf(m);
}

// ---------------- launch ----------------

extern "C" void kernel_launch(void* const* d_in, const int* in_sizes, int n_in,
                              void* d_out, int out_size, void* d_ws, size_t ws_size,
                              hipStream_t stream) {
  const float* x   = (const float*)d_in[0];  // (64,3,4096)
  const float* shp = (const float*)d_in[1];  // (3,128,64)
  float* out = (float*)d_out;                // (64,1,128)

  char* ws = (char*)d_ws;
  size_t off = 0;
  auto take = [&](size_t bytes) -> char* {
    char* p = ws + off;
    off = (off + bytes + 255) & ~(size_t)255;
    return p;
  };
  unsigned short* xh   = (unsigned short*)take((size_t)N_ * C_ * LXPAD_ * 2);
  unsigned short* xl   = (unsigned short*)take((size_t)N_ * C_ * LXPAD_ * 2);
  unsigned short* shh  = (unsigned short*)take((size_t)C_ * K_ * S_ * 2);
  unsigned short* shl  = (unsigned short*)take((size_t)C_ * K_ * S_ * 2);
  float*          s2   = (float*)take((size_t)C_ * K_ * 4);
  float*          x2w  = (float*)take((size_t)N_ * C_ * WPAD_ * 4);
  float*          part = (float*)take((size_t)N_ * 8 * C_ * 4 * 16 * 4);
  (void)ws_size; (void)in_sizes; (void)n_in; (void)out_size;

  {
    int tot = N_ * C_ * LXPAD_;
    cvt_x_kernel<<<(tot + 255) / 256, 256, 0, stream>>>(x, xh, xl);
  }
  cvt_s_kernel<<<(C_ * K_ + 127) / 128, 128, 0, stream>>>(shp, shh, shl, s2);
  {
    int tot = N_ * C_ * WPAD_;
    x2win_kernel<<<(tot + 255) / 256, 256, 0, stream>>>(x, x2w);
  }
  {
    dim3 grid(N_, 4, C_);
    shapelet_main_kernel<<<grid, 128, 0, stream>>>(xh, xl, shh, shl, s2, x2w, part);
  }
  reduce_kernel<<<N_, K_, 0, stream>>>(part, out);
}
